// ContrastiveWSD_42365557407851
// MI455X (gfx1250) — matrix operand order repaired and checked by reading, not compiled, not run on hardware
//
#include <hip/hip_runtime.h>
#include <hip/hip_bf16.h>
#include <math.h>

#define HD 768
#define NNODES 50000
#define NEDGE_RAW 100000
#define NEDGE 150000
#define NEG_SLOPE 0.2f
#define KSTEPS (HD / 32)

typedef __attribute__((ext_vector_type(16))) _Float16 v16h;
typedef __attribute__((ext_vector_type(8)))  float    v8f;

// ---------------------------------------------------------------------------
// LayerNorm + exact GELU. Writes f16 (GEMM operand) and/or f32.
// One block (256 threads) per row; 768 = 3 * 256.
// ---------------------------------------------------------------------------
__global__ __launch_bounds__(256) void ln_gelu_k(
    const float* __restrict__ X, const float* __restrict__ g,
    const float* __restrict__ b, _Float16* __restrict__ outH,
    float* __restrict__ outF)
{
    __shared__ float red[256];
    __shared__ float s_mu, s_rstd;
    const int row = blockIdx.x;
    const int tid = threadIdx.x;

    float v[3];
    float s = 0.f, s2 = 0.f;
    #pragma unroll
    for (int j = 0; j < 3; ++j) {
        float x = X[(size_t)row * HD + tid + j * 256];
        v[j] = x; s += x; s2 += x * x;
    }
    red[tid] = s; __syncthreads();
    for (int o = 128; o; o >>= 1) { if (tid < o) red[tid] += red[tid + o]; __syncthreads(); }
    if (tid == 0) s_mu = red[0] * (1.0f / HD);
    __syncthreads();
    red[tid] = s2; __syncthreads();
    for (int o = 128; o; o >>= 1) { if (tid < o) red[tid] += red[tid + o]; __syncthreads(); }
    if (tid == 0) {
        float mu = s_mu;
        float var = red[0] * (1.0f / HD) - mu * mu;
        s_rstd = rsqrtf(var + 1e-5f);
    }
    __syncthreads();
    const float mu = s_mu, rstd = s_rstd;
    #pragma unroll
    for (int j = 0; j < 3; ++j) {
        int k = tid + j * 256;
        float y = (v[j] - mu) * rstd * g[k] + b[k];
        float z = 0.5f * y * (1.0f + erff(y * 0.70710678118654752f));
        if (outH) outH[(size_t)row * HD + k] = (_Float16)z;
        if (outF) outF[(size_t)row * HD + k] = z;
    }
}

// ---------------------------------------------------------------------------
// f32 -> f16 conversion with transpose: Wt[n][k] = (f16) W[k][n]  (768x768)
// Output index coalesced; strided f32 reads (4.7 MB once per weight, noise).
// ---------------------------------------------------------------------------
__global__ __launch_bounds__(256) void f2h_t_k(const float* __restrict__ W,
                                               _Float16* __restrict__ Wt)
{
    int i = blockIdx.x * 256 + threadIdx.x;   // output index, n*768 + k
    if (i < HD * HD) {
        int n = i / HD;
        int k = i - n * HD;
        Wt[i] = (_Float16)W[k * HD + n];
    }
}

// ---------------------------------------------------------------------------
// GEMM: Y[M,768] = Xh[M,768] @ W[768,768] + bias, f16 in / f32 out.
// W supplied pre-transposed (Wt[n][k]) so both tiles stage with b128 LDS ops.
// Block: 256 thr = 8 waves. Tile 128(M) x 64(N), K-step 32, double-buffered
// LDS with register staging to overlap global loads with the WMMAs.
// ---------------------------------------------------------------------------
__global__ __launch_bounds__(256) void gemm_bias_k(
    const _Float16* __restrict__ Xh, const _Float16* __restrict__ Wt,
    const float* __restrict__ bias, float* __restrict__ Y, int M)
{
    __shared__ _Float16 lA[2][128][32];   // row-major A tile
    __shared__ _Float16 lB[2][64][32];    // Bt[n][k]

    const int tid  = threadIdx.x;
    const int lane = tid & 31;
    const int w    = tid >> 5;         // wave id 0..7
    const int l16  = lane & 15;
    const int hi   = lane >> 4;
    const int blockM = blockIdx.x * 128;
    const int n0     = blockIdx.y * 64;

    // staging assignments
    const int  arow  = tid >> 1;                 // 0..127
    const int  akoff = (tid & 1) * 16;           // 0 or 16 halves
    const bool aval  = (blockM + arow) < M;
    const _Float16* aptr = Xh + (size_t)(blockM + arow) * HD + akoff;

    const bool doB   = tid < 128;
    const int  brow  = tid >> 1;                 // 0..63 (when doB)
    const int  bkoff = (tid & 1) * 16;
    const _Float16* bptr = Wt + (size_t)(n0 + brow) * HD + bkoff;

    v8f acc[4];
    #pragma unroll
    for (int i = 0; i < 4; ++i)
        #pragma unroll
        for (int j = 0; j < 8; ++j) acc[i][j] = 0.f;

    // ---- prologue: fetch + commit tile 0
    uint4 ra0 = make_uint4(0u,0u,0u,0u), ra1 = make_uint4(0u,0u,0u,0u);
    uint4 rb0 = make_uint4(0u,0u,0u,0u), rb1 = make_uint4(0u,0u,0u,0u);
    if (aval) { ra0 = ((const uint4*)aptr)[0]; ra1 = ((const uint4*)aptr)[1]; }
    if (doB)  { rb0 = ((const uint4*)bptr)[0]; rb1 = ((const uint4*)bptr)[1]; }
    *(uint4*)&lA[0][arow][akoff]     = ra0;
    *(uint4*)&lA[0][arow][akoff + 8] = ra1;
    if (doB) {
        *(uint4*)&lB[0][brow][bkoff]     = rb0;
        *(uint4*)&lB[0][brow][bkoff + 8] = rb1;
    }
    __syncthreads();

    int cur = 0;
    for (int kt = 0; kt < KSTEPS; ++kt) {
        // ---- issue global loads for tile kt+1 (overlaps with WMMAs below)
        if (kt + 1 < KSTEPS) {
            const int kn = (kt + 1) * 32;
            if (aval) {
                ra0 = ((const uint4*)(aptr + kn))[0];
                ra1 = ((const uint4*)(aptr + kn))[1];
            }
            if (doB) {
                rb0 = ((const uint4*)(bptr + kn))[0];
                rb1 = ((const uint4*)(bptr + kn))[1];
            }
        }

        // ---- compute from current buffer
        v16h afr;
        ((uint4*)&afr)[0] = *(const uint4*)&lA[cur][w * 16 + l16][hi * 8];
        ((uint4*)&afr)[1] = *(const uint4*)&lA[cur][w * 16 + l16][16 + hi * 8];
        #pragma unroll
        for (int nt = 0; nt < 4; ++nt) {
            v16h bfr;
            const uint4* pb = (const uint4*)&lB[cur][nt * 16 + l16][hi * 8];
            ((uint4*)&bfr)[0] = pb[0];
            ((uint4*)&bfr)[1] = pb[2];
            acc[nt] = __builtin_amdgcn_wmma_f32_16x16x32_f16(
                false, afr, false, bfr, (short)0, acc[nt], false, false);
        }

        // ---- commit staged registers to the alternate buffer
        if (kt + 1 < KSTEPS) {
            const int nxt = cur ^ 1;
            *(uint4*)&lA[nxt][arow][akoff]     = ra0;
            *(uint4*)&lA[nxt][arow][akoff + 8] = ra1;
            if (doB) {
                *(uint4*)&lB[nxt][brow][bkoff]     = rb0;
                *(uint4*)&lB[nxt][brow][bkoff + 8] = rb1;
            }
        }
        __syncthreads();
        cur ^= 1;
    }

    // ---- store: VGPR r -> row hi*8 + r, col = l16 (ISA C/D layout)
    #pragma unroll
    for (int nt = 0; nt < 4; ++nt) {
        int col = n0 + nt * 16 + l16;
        float bv = bias[col];
        #pragma unroll
        for (int r = 0; r < 8; ++r) {
            int row = blockM + w * 16 + hi * 8 + r;
            if (row < M) Y[(size_t)row * HD + col] = acc[nt][r] + bv;
        }
    }
}

// ---------------------------------------------------------------------------
// Segment-softmax scaffolding
// ---------------------------------------------------------------------------
__device__ __forceinline__ void atomicMaxF(float* addr, float val)
{
    unsigned* a = (unsigned*)addr;
    unsigned old = *a;
    while (val > __uint_as_float(old)) {
        unsigned assumed = old;
        old = atomicCAS(a, assumed, __float_as_uint(val));
        if (old == assumed) break;
    }
}

__global__ __launch_bounds__(256) void init_md_k(float* m, float* denom)
{
    int i = blockIdx.x * 256 + threadIdx.x;
    if (i < NNODES) { m[i] = -3.0e38f; denom[i] = 0.f; }
}

__global__ __launch_bounds__(256) void init_out_k(float* __restrict__ out,
                                                  const float* __restrict__ bias)
{
    size_t i = (size_t)blockIdx.x * 256 + threadIdx.x;
    if (i < (size_t)NNODES * HD) out[i] = bias[i % HD];
}

// one wave per edge: e = leakyrelu(xl[src]+xr[dst]) . att ; segment max
__global__ __launch_bounds__(256) void edge_score_k(
    const float* __restrict__ xl, const float* __restrict__ xr,
    const int* __restrict__ edges, const float* __restrict__ att,
    float* __restrict__ esc, float* __restrict__ m)
{
    int e = blockIdx.x * 8 + (threadIdx.x >> 5);
    int lane = threadIdx.x & 31;
    if (e >= NEDGE) return;
    int src, dst;
    if (e < NEDGE_RAW) { src = edges[e]; dst = edges[NEDGE_RAW + e]; }
    else               { src = dst = e - NEDGE_RAW; }
    const float* pl = xl + (size_t)src * HD;
    const float* pr = xr + (size_t)dst * HD;
    float s = 0.f;
    #pragma unroll
    for (int j = 0; j < HD / 32; ++j) {
        int k = lane + j * 32;
        float v = pl[k] + pr[k];
        v = v > 0.f ? v : NEG_SLOPE * v;
        s += v * att[k];
    }
    #pragma unroll
    for (int off = 16; off; off >>= 1) s += __shfl_down(s, off, 32);
    if (lane == 0) { esc[e] = s; atomicMaxF(&m[dst], s); }
}

// one thread per edge: a = exp(e - m[dst]); denom[dst] += a
__global__ __launch_bounds__(256) void edge_exp_k(
    float* __restrict__ esc, const float* __restrict__ m,
    float* __restrict__ denom, const int* __restrict__ edges)
{
    int e = blockIdx.x * 256 + threadIdx.x;
    if (e >= NEDGE) return;
    int dst = (e < NEDGE_RAW) ? edges[NEDGE_RAW + e] : e - NEDGE_RAW;
    float a = expf(esc[e] - m[dst]);
    esc[e] = a;
    atomicAdd(&denom[dst], a);
}

// one wave per edge: out[dst] += (a/denom[dst]) * xl[src]
__global__ __launch_bounds__(256) void edge_scatter_k(
    const float* __restrict__ esc, const float* __restrict__ denom,
    const float* __restrict__ xl, const int* __restrict__ edges,
    float* __restrict__ out)
{
    int e = blockIdx.x * 8 + (threadIdx.x >> 5);
    int lane = threadIdx.x & 31;
    if (e >= NEDGE) return;
    int src, dst;
    if (e < NEDGE_RAW) { src = edges[e]; dst = edges[NEDGE_RAW + e]; }
    else               { src = dst = e - NEDGE_RAW; }
    float alpha = esc[e] / denom[dst];
    const float* pl = xl + (size_t)src * HD;
    float* po = out + (size_t)dst * HD;
    #pragma unroll
    for (int j = 0; j < HD / 32; ++j) {
        int k = lane + j * 32;
        atomicAdd(&po[k], alpha * pl[k]);
    }
}

__global__ __launch_bounds__(256) void copy_f32_k(const float* __restrict__ src,
                                                  float* __restrict__ dst, int n)
{
    int i = blockIdx.x * 256 + threadIdx.x;
    if (i < n) dst[i] = src[i];
}

// ---------------------------------------------------------------------------
extern "C" void kernel_launch(void* const* d_in, const int* in_sizes, int n_in,
                              void* d_out, int out_size, void* d_ws, size_t ws_size,
                              hipStream_t stream)
{
    (void)in_sizes; (void)n_in; (void)out_size; (void)ws_size;

    const float* emb        = (const float*)d_in[0];
    const float* node_feats = (const float*)d_in[1];
    const int*   edges      = (const int*)d_in[2];
    // d_in[3] = labels_size (4096), baked into launch geometry
    const float* g2_Wl  = (const float*)d_in[4];
    const float* g2_bl  = (const float*)d_in[5];
    const float* g2_Wr  = (const float*)d_in[6];
    const float* g2_br  = (const float*)d_in[7];
    const float* g2_att = (const float*)d_in[8];
    const float* g2_bias= (const float*)d_in[9];
    const float* g3_Wl  = (const float*)d_in[10];
    const float* g3_bl  = (const float*)d_in[11];
    const float* g3_Wr  = (const float*)d_in[12];
    const float* g3_br  = (const float*)d_in[13];
    const float* g3_att = (const float*)d_in[14];
    const float* g3_bias= (const float*)d_in[15];
    const float* ln1_g  = (const float*)d_in[16];
    const float* ln1_b  = (const float*)d_in[17];
    const float* ln2_g  = (const float*)d_in[18];
    const float* ln2_b  = (const float*)d_in[19];
    const float* ln3_g  = (const float*)d_in[20];
    const float* ln3_b  = (const float*)d_in[21];

    float* out_f = (float*)d_out;

    // workspace partition
    char* p = (char*)d_ws;
    _Float16* Xh  = (_Float16*)p; p += (size_t)NNODES * HD * 2;
    _Float16* Wlh = (_Float16*)p; p += (size_t)HD * HD * 2;   // transposed f16
    _Float16* Wrh = (_Float16*)p; p += (size_t)HD * HD * 2;   // transposed f16
    float* xl    = (float*)p; p += (size_t)NNODES * HD * 4;
    float* xr    = (float*)p; p += (size_t)NNODES * HD * 4;   // reused as GAT out
    float* esc   = (float*)p; p += (size_t)NEDGE * 4;
    float* mbuf  = (float*)p; p += (size_t)NNODES * 4;
    float* denom = (float*)p; p += (size_t)NNODES * 4;
    float* gat_out = xr;   // alias: xr dead after edge_score

    const dim3 gemm_grid((NNODES + 127) / 128, HD / 64);
    const int ew_blocks = (NEDGE + 7) / 8;
    const int et_blocks = (NEDGE + 255) / 256;
    const int init_out_blocks = (int)(((size_t)NNODES * HD + 255) / 256);
    const int wconv_blocks = (HD * HD + 255) / 256;

    // ---------------- layer 1 LN+GELU ----------------
    ln_gelu_k<<<NNODES, 256, 0, stream>>>(node_feats, ln1_g, ln1_b, Xh, nullptr);

    // ---------------- GAT layer 2 ----------------
    f2h_t_k<<<wconv_blocks, 256, 0, stream>>>(g2_Wl, Wlh);
    f2h_t_k<<<wconv_blocks, 256, 0, stream>>>(g2_Wr, Wrh);
    gemm_bias_k<<<gemm_grid, 256, 0, stream>>>(Xh, Wlh, g2_bl, xl, NNODES);
    gemm_bias_k<<<gemm_grid, 256, 0, stream>>>(Xh, Wrh, g2_br, xr, NNODES);
    init_md_k<<<(NNODES + 255) / 256, 256, 0, stream>>>(mbuf, denom);
    edge_score_k<<<ew_blocks, 256, 0, stream>>>(xl, xr, edges, g2_att, esc, mbuf);
    edge_exp_k<<<et_blocks, 256, 0, stream>>>(esc, mbuf, denom, edges);
    init_out_k<<<init_out_blocks, 256, 0, stream>>>(gat_out, g2_bias);   // xr now dead
    edge_scatter_k<<<ew_blocks, 256, 0, stream>>>(esc, denom, xl, edges, gat_out);

    // ---------------- LN+GELU 2 ----------------
    ln_gelu_k<<<NNODES, 256, 0, stream>>>(gat_out, ln2_g, ln2_b, Xh, nullptr);

    // ---------------- GAT layer 3 ----------------
    f2h_t_k<<<wconv_blocks, 256, 0, stream>>>(g3_Wl, Wlh);
    f2h_t_k<<<wconv_blocks, 256, 0, stream>>>(g3_Wr, Wrh);
    gemm_bias_k<<<gemm_grid, 256, 0, stream>>>(Xh, Wlh, g3_bl, xl, NNODES);
    gemm_bias_k<<<gemm_grid, 256, 0, stream>>>(Xh, Wrh, g3_br, xr, NNODES);
    init_md_k<<<(NNODES + 255) / 256, 256, 0, stream>>>(mbuf, denom);
    edge_score_k<<<ew_blocks, 256, 0, stream>>>(xl, xr, edges, g3_att, esc, mbuf);
    edge_exp_k<<<et_blocks, 256, 0, stream>>>(esc, mbuf, denom, edges);
    init_out_k<<<init_out_blocks, 256, 0, stream>>>(gat_out, g3_bias);
    edge_scatter_k<<<ew_blocks, 256, 0, stream>>>(esc, denom, xl, edges, gat_out);

    // ---------------- final LN+GELU: only first 4096 rows, straight to d_out ----
    ln_gelu_k<<<4096, 256, 0, stream>>>(gat_out, ln3_g, ln3_b, nullptr,
                                        out_f + 32 * HD);

    // ---------------- output[0]: input_embeddings passthrough ----------------
    copy_f32_k<<<(32 * HD + 255) / 256, 256, 0, stream>>>(emb, out_f, 32 * HD);
}
